// Model_81956565942348
// MI455X (gfx1250) — compile-verified, matrix operand order checked by reading, not run on hardware
//
#include <hip/hip_runtime.h>
#include <hip/hip_bf16.h>

typedef __attribute__((ext_vector_type(16))) __bf16 v16bf;
typedef __attribute__((ext_vector_type(8)))  float  v8f;

#define N_NODES 50000
#define N_EDGES 1048576
#define NODE_H  128
#define EDGE_H  64
#define GNN_H   128
#define K1      (2*NODE_H + EDGE_H)   // 320
#define K2      (NODE_H + GNN_H)      // 256
#define S1      (K1 + 8)              // 328 bf16: padded K-stride, bank-conflict-free
#define S2      (K2 + 8)              // 264 bf16

// ---------------------------------------------------------------- utilities

__global__ void zero_f32_kernel(float* __restrict__ p, long long n) {
    long long i = (long long)blockIdx.x * blockDim.x + threadIdx.x;
    long long stride = (long long)gridDim.x * blockDim.x;
    for (; i < n; i += stride) p[i] = 0.0f;
}

__global__ void cvt_f32_to_bf16(const float* __restrict__ src, __bf16* __restrict__ dst, long long n) {
    long long i = (long long)blockIdx.x * blockDim.x + threadIdx.x;
    long long stride = (long long)gridDim.x * blockDim.x;
    for (; i < n; i += stride) dst[i] = (__bf16)src[i];
}

// w: [K][N] row-major f32  ->  wt: [N][S] bf16 (K-stride padded to S)
__global__ void transpose_to_bf16(const float* __restrict__ w, __bf16* __restrict__ wt,
                                  int K, int N, int S) {
    int i = blockIdx.x * blockDim.x + threadIdx.x;
    int total = K * N;
    int stride = gridDim.x * blockDim.x;
    for (; i < total; i += stride) {
        int n = i / K;
        int k = i - n * K;
        wt[(long long)n * S + k] = (__bf16)w[(long long)k * N + n];
    }
}

// 8 contiguous bf16 (16B)
__device__ __forceinline__ void fill8_bf(const __bf16* __restrict__ p, __bf16* d) {
    *(uint4*)d = *(const uint4*)p;
}
// 8 contiguous f32 -> 8 bf16
__device__ __forceinline__ void fill8_f32(const float* __restrict__ p, __bf16* d) {
    float4 a = ((const float4*)p)[0];
    float4 b = ((const float4*)p)[1];
    d[0] = (__bf16)a.x; d[1] = (__bf16)a.y; d[2] = (__bf16)a.z; d[3] = (__bf16)a.w;
    d[4] = (__bf16)b.x; d[5] = (__bf16)b.y; d[6] = (__bf16)b.z; d[7] = (__bf16)b.w;
}
// 8 contiguous f32 * scale -> 8 bf16
__device__ __forceinline__ void fill8_f32_scaled(const float* __restrict__ p, float s, __bf16* d) {
    float4 a = ((const float4*)p)[0];
    float4 b = ((const float4*)p)[1];
    d[0] = (__bf16)(a.x*s); d[1] = (__bf16)(a.y*s); d[2] = (__bf16)(a.z*s); d[3] = (__bf16)(a.w*s);
    d[4] = (__bf16)(b.x*s); d[5] = (__bf16)(b.y*s); d[6] = (__bf16)(b.z*s); d[7] = (__bf16)(b.w*s);
}

// build one A fragment (row m of [x[r] | x[c] | ea[e]]) for K-step starting at k0
__device__ __forceinline__ v16bf build_afrag(const __bf16* __restrict__ xb,
                                             const float* __restrict__ ea,
                                             int r, int c, int e, int k0) {
    __align__(32) __bf16 afrag[16];
    #pragma unroll
    for (int run = 0; run < 2; ++run) {
        const int k = k0 + run * 16;
        __bf16* d = afrag + run * 8;
        if (k < NODE_H)
            fill8_bf(xb + (long long)r * NODE_H + k, d);
        else if (k < 2 * NODE_H)
            fill8_bf(xb + (long long)c * NODE_H + (k - NODE_H), d);
        else
            fill8_f32(ea + (long long)e * EDGE_H + (k - 2 * NODE_H), d);
    }
    return *(v16bf*)afrag;
}

extern __shared__ __align__(16) char smem_raw[];

// ---------------------------------------------------------------- edge MLP + scatter
// Each wave: 32 edges x 128 outputs. Every LDS B fragment feeds TWO WMMAs
// (halves LDS bandwidth per edge); B loads grouped 4-wide ahead of the WMMA
// burst via sched_barrier so the matrix pipe issues back-to-back.
__global__ __launch_bounds__(256) void edge_mlp_scatter(
    const __bf16* __restrict__ xb,      // [N_NODES][128] bf16
    const int*    __restrict__ rowi,    // [E]
    const int*    __restrict__ coli,    // [E]
    const float*  __restrict__ ea,      // [E][64] f32
    const __bf16* __restrict__ w1t,     // [128][S1] bf16 (N-major, padded)
    const float*  __restrict__ b1,      // [128]
    float* __restrict__ s,              // [N_NODES][128] accumulator
    float* __restrict__ cnt)            // [N_NODES]
{
    __bf16* sW = (__bf16*)smem_raw;     // 128*S1 bf16 = 82KB
    {
        const uint4* src = (const uint4*)w1t;
        uint4* dst = (uint4*)sW;
        for (int i = threadIdx.x; i < (GNN_H * S1) / 8; i += blockDim.x) dst[i] = src[i];
    }
    __syncthreads();

    const int lane = threadIdx.x & 31;
    const int wid  = threadIdx.x >> 5;
    const int lh   = lane >> 4;        // lane half: K-shift of 8
    const int lm   = lane & 15;        // row (A) / col (B,C)
    const int nPairs  = N_EDGES / 32;  // 32768 double-tiles
    const int pStride = gridDim.x * 8;

    // bias is tile-invariant: hoist
    float bias[8];
    #pragma unroll
    for (int nt = 0; nt < 8; ++nt) bias[nt] = b1[nt * 16 + lm];

    for (int pair = blockIdx.x * 8 + wid; pair < nPairs; pair += pStride) {
        const int e0 = pair * 32 + lm;       // sub-tile 0, row lm
        const int e1 = e0 + 16;              // sub-tile 1, row lm
        const int r0 = rowi[e0], c0 = coli[e0];
        const int r1 = rowi[e1], c1 = coli[e1];

        // prefetch next pair's gathered rows one K-loop ahead
        {
            const int pn  = pair + pStride;
            const int en0 = (pn < nPairs) ? pn * 32 + lm : e0;
            const int rn0 = rowi[en0], cn0 = coli[en0];
            const int rn1 = rowi[en0 + 16], cn1 = coli[en0 + 16];
            __builtin_prefetch(xb + (long long)rn0 * NODE_H, 0, 1);
            __builtin_prefetch(xb + (long long)cn0 * NODE_H, 0, 1);
            __builtin_prefetch(xb + (long long)rn1 * NODE_H, 0, 1);
            __builtin_prefetch(xb + (long long)cn1 * NODE_H, 0, 1);
            __builtin_prefetch(ea + (long long)en0 * EDGE_H, 0, 1);
        }

        v8f acc0[8] = {};
        v8f acc1[8] = {};

        for (int ks = 0; ks < K1 / 32; ++ks) {
            const int k0 = ks * 32 + lh * 8;
            const v16bf av0 = build_afrag(xb, ea, r0, c0, e0, k0);
            const v16bf av1 = build_afrag(xb, ea, r1, c1, e1, k0);

            #pragma unroll
            for (int g = 0; g < 2; ++g) {
                v16bf bv[4];
                #pragma unroll
                for (int i = 0; i < 4; ++i) {
                    const int nt = g * 4 + i;
                    const __bf16* bp = sW + (nt * 16 + lm) * S1 + k0;
                    __align__(32) __bf16 bfrag[16];
                    *(uint4*)(bfrag)     = *(const uint4*)(bp);
                    *(uint4*)(bfrag + 8) = *(const uint4*)(bp + 16);
                    bv[i] = *(v16bf*)bfrag;
                }
                __builtin_amdgcn_sched_barrier(0);   // keep loads ahead of WMMA burst
                #pragma unroll
                for (int i = 0; i < 4; ++i) {
                    const int nt = g * 4 + i;
                    acc0[nt] = __builtin_amdgcn_wmma_f32_16x16x32_bf16(
                        false, av0, false, bv[i], (short)0, acc0[nt], false, false);
                    acc1[nt] = __builtin_amdgcn_wmma_f32_16x16x32_bf16(
                        false, av1, false, bv[i], (short)0, acc1[nt], false, false);
                }
            }
        }

        // scatter column indices live in other lanes -> shuffle, no reloads
        int cn0_[8], cn1_[8];
        #pragma unroll
        for (int j = 0; j < 8; ++j) {
            cn0_[j] = __shfl(c0, j + lh * 8, 32);
            cn1_[j] = __shfl(c1, j + lh * 8, 32);
        }

        // bias + ReLU + atomic scatter into s[col]
        #pragma unroll
        for (int nt = 0; nt < 8; ++nt) {
            const int n = nt * 16 + lm;
            #pragma unroll
            for (int j = 0; j < 8; ++j) {
                float v0 = acc0[nt][j] + bias[nt];
                v0 = v0 > 0.0f ? v0 : 0.0f;
                atomicAdd(s + (long long)cn0_[j] * GNN_H + n, v0);
                float v1 = acc1[nt][j] + bias[nt];
                v1 = v1 > 0.0f ? v1 : 0.0f;
                atomicAdd(s + (long long)cn1_[j] * GNN_H + n, v1);
            }
        }
        if (lane < 16) {
            atomicAdd(cnt + c0, 1.0f);
            atomicAdd(cnt + c1, 1.0f);
        }
    }
}

// ---------------------------------------------------------------- node MLP
// out = relu([x | s/max(cnt,1)] @ W2 + b2); each wave: 16 nodes x 128 outputs.
__global__ __launch_bounds__(256) void node_mlp(
    const __bf16* __restrict__ xb,      // [N_NODES][128] bf16
    const float*  __restrict__ s,       // [N_NODES][128]
    const float*  __restrict__ cnt,     // [N_NODES]
    const __bf16* __restrict__ w2t,     // [128][S2] bf16 (N-major, padded)
    const float*  __restrict__ b2,      // [128]
    float* __restrict__ out)            // [N_NODES][128]
{
    __bf16* sW = (__bf16*)smem_raw;     // 128*S2 bf16 = 66KB
    {
        const uint4* src = (const uint4*)w2t;
        uint4* dst = (uint4*)sW;
        for (int i = threadIdx.x; i < (GNN_H * S2) / 8; i += blockDim.x) dst[i] = src[i];
    }
    __syncthreads();

    const int lane = threadIdx.x & 31;
    const int wid  = threadIdx.x >> 5;
    const int lh   = lane >> 4;
    const int lm   = lane & 15;
    const int nTiles = N_NODES / 16;    // 3125 exactly

    float bias[8];
    #pragma unroll
    for (int nt = 0; nt < 8; ++nt) bias[nt] = b2[nt * 16 + lm];

    for (int tile = blockIdx.x * 8 + wid; tile < nTiles; tile += gridDim.x * 8) {
        const int node = tile * 16 + lm;
        const float cinv = 1.0f / fmaxf(cnt[node], 1.0f);

        v8f acc[8] = {};

        for (int ks = 0; ks < K2 / 32; ++ks) {
            const int k0 = ks * 32 + lh * 8;
            __align__(32) __bf16 afrag[16];
            #pragma unroll
            for (int run = 0; run < 2; ++run) {
                const int k = k0 + run * 16;
                __bf16* d = afrag + run * 8;
                if (k < NODE_H)
                    fill8_bf(xb + (long long)node * NODE_H + k, d);
                else
                    fill8_f32_scaled(s + (long long)node * GNN_H + (k - NODE_H), cinv, d);
            }
            v16bf av = *(v16bf*)afrag;

            v16bf bv[8];
            #pragma unroll
            for (int nt = 0; nt < 8; ++nt) {
                const __bf16* bp = sW + (nt * 16 + lm) * S2 + k0;
                __align__(32) __bf16 bfrag[16];
                *(uint4*)(bfrag)     = *(const uint4*)(bp);
                *(uint4*)(bfrag + 8) = *(const uint4*)(bp + 16);
                bv[nt] = *(v16bf*)bfrag;
            }
            __builtin_amdgcn_sched_barrier(0);       // loads first, then WMMA burst
            #pragma unroll
            for (int nt = 0; nt < 8; ++nt)
                acc[nt] = __builtin_amdgcn_wmma_f32_16x16x32_bf16(
                    false, av, false, bv[nt], (short)0, acc[nt], false, false);
        }

        #pragma unroll
        for (int nt = 0; nt < 8; ++nt) {
            const int n = nt * 16 + lm;
            #pragma unroll
            for (int j = 0; j < 8; ++j) {
                const int m = j + lh * 8;
                const int onode = tile * 16 + m;
                float v = acc[nt][j] + bias[nt];
                v = v > 0.0f ? v : 0.0f;
                out[(long long)onode * GNN_H + n] = v;
            }
        }
    }
}

// ---------------------------------------------------------------- host launch

extern "C" void kernel_launch(void* const* d_in, const int* in_sizes, int n_in,
                              void* d_out, int out_size, void* d_ws, size_t ws_size,
                              hipStream_t stream) {
    const float* x   = (const float*)d_in[0];
    const int*   ei  = (const int*)d_in[1];     // [2, E]
    const float* ea  = (const float*)d_in[2];
    const float* W1  = (const float*)d_in[3];   // [320,128]
    const float* b1  = (const float*)d_in[4];
    const float* W2  = (const float*)d_in[5];   // [256,128]
    const float* b2  = (const float*)d_in[6];
    float* out = (float*)d_out;

    const int* rowi = ei;
    const int* coli = ei + N_EDGES;

    // workspace carve-out (256B aligned): s | cnt | x_bf16 | W1T | W2T  (~38.9 MB)
    char* ws = (char*)d_ws;
    size_t off = 0;
    auto carve = [&](size_t bytes) -> void* {
        void* p = ws + off;
        off += (bytes + 255) & ~(size_t)255;
        return p;
    };
    float*  sAcc = (float*)carve((size_t)N_NODES * GNN_H * sizeof(float));
    float*  cnt  = (float*)carve((size_t)N_NODES * sizeof(float));
    __bf16* xb   = (__bf16*)carve((size_t)N_NODES * NODE_H * sizeof(__bf16));
    __bf16* w1t  = (__bf16*)carve((size_t)GNN_H * S1 * sizeof(__bf16));
    __bf16* w2t  = (__bf16*)carve((size_t)GNN_H * S2 * sizeof(__bf16));

    // 1) zero accumulator + counts (s and cnt are contiguous: s size is 256B-multiple)
    zero_f32_kernel<<<2048, 256, 0, stream>>>(sAcc, (long long)N_NODES * GNN_H + N_NODES);

    // 2) precision conversions / weight transposes (padded K-stride)
    cvt_f32_to_bf16<<<4096, 256, 0, stream>>>(x, xb, (long long)N_NODES * NODE_H);
    transpose_to_bf16<<<(K1 * GNN_H + 255) / 256, 256, 0, stream>>>(W1, w1t, K1, GNN_H, S1);
    transpose_to_bf16<<<(K2 * GNN_H + 255) / 256, 256, 0, stream>>>(W2, w2t, K2, GNN_H, S2);

    // 3) edge MLP (bf16 WMMA, 32-edge M-tiles) + atomic scatter-sum; W1T in 82KB LDS
    edge_mlp_scatter<<<1024, 256, GNN_H * S1 * sizeof(__bf16), stream>>>(
        xb, rowi, coli, ea, w1t, b1, sAcc, cnt);

    // 4) node MLP (bf16 WMMA); padded W2T staged in 66KB dynamic LDS
    node_mlp<<<(N_NODES / 16 + 7) / 8, 256, GNN_H * S2 * sizeof(__bf16), stream>>>(
        xb, sAcc, cnt, w2t, b2, out);
}
